// VectorQuantizer_64355789963295
// MI455X (gfx1250) — matrix operand order repaired and checked by reading, not compiled, not run on hardware
//
#include <hip/hip_runtime.h>
#include <hip/hip_bf16.h>

typedef __attribute__((ext_vector_type(16))) _Float16 v16h;
typedef __attribute__((ext_vector_type(8)))  float    v8f;

#define C_DIM     64
#define K_CODES   512
#define S_DIM     16384            // T*H*W = 16*32*32
#define N_ROWS    262144           // B*S = 16*16384
#define Q_ELEMS   16777216         // B*C*T*H*W
#define LOSS_OFF  Q_ELEMS
#define IDX_OFF   (Q_ELEMS + 1)
#define EMB_SCALE 512.0f

union V16H { v16h v; _Float16 h[16]; };
union V8F  { v8f  v; float    f[8];  };

// ds_swizzle group-of-32 XOR mode: lane reads lane^XM (single DS instruction,
// no exec-mask gymnastics, EXEC is all-ones at every call site).
template <int XM>
__device__ __forceinline__ int swz_i(int x) {
  return __builtin_amdgcn_ds_swizzle(x, (XM << 10) | 0x1f);
}
template <int XM>
__device__ __forceinline__ float swz_f(float x) {
  return __int_as_float(__builtin_amdgcn_ds_swizzle(__float_as_int(x), (XM << 10) | 0x1f));
}

__global__ __launch_bounds__(128) void vq_wmma_kernel(const float* __restrict__ z,
                                                      const float* __restrict__ emb,
                                                      float* __restrict__ out) {
  // 64 KB codebook staged as scaled fp16 in WMMA-B per-lane layout + f32 bias table
  __shared__ alignas(16) _Float16 sB[32768];
  __shared__ float sBias[K_CODES];
  __shared__ int   sIdx[64];

  const int t = threadIdx.x;

  // ---- stage codebook into LDS ----
  // element i -> (g = i>>10, hk = (i>>9)&1, lane = (i>>4)&31, j = i&15)
  // B as staged here (contiguous k, same order as A): lanes 0-15 hold K=0..15, lanes 16-31 hold K=16..31
  for (int i = t; i < 32768; i += 128) {
    int j    = i & 15;
    int ln   = (i >> 4) & 31;
    int hk   = (i >> 9) & 1;
    int g    = i >> 10;
    int lh   = ln >> 4;
    int c    = j + 16 * lh + 32 * hk;       // channel (K index)
    int code = g * 16 + (ln & 15);          // column (N index)
    sB[i] = (_Float16)(EMB_SCALE * emb[code * C_DIM + c]);
  }
  for (int k = t; k < K_CODES; k += 128) {
    const float* e = emb + k * C_DIM;
    float s = 0.f;
    for (int c = 0; c < C_DIM; ++c) s += e[c] * e[c];
    sBias[k] = -0.5f * EMB_SCALE * s;       // argmax(z.e - 0.5|e|^2), scaled by 512
  }
  __syncthreads();

  const int wid  = t >> 5;
  const int lane = t & 31;
  const int m    = lane & 15;               // row (A) / column (B,C) sub-index
  const int lh   = lane >> 4;

  const int tile = blockIdx.x * 4 + wid;    // 16384 tiles of 16 rows
  const int n0   = tile * 16;
  const int b    = n0 >> 14;                // n0 / S_DIM
  const int s0   = n0 & (S_DIM - 1);
  const float* zb   = z + (size_t)b * C_DIM * S_DIM;
  const float* zrow = zb + s0 + m;

  // ---- load A tile (16 rows x 64 channels) as two fp16 K=32 fragments ----
  // A as loaded here (contiguous k, same order as B): lanes 0-15: V0..7 -> K 0..15, lanes 16-31: V0..7 -> K 16..31
  V16H a0, a1;
#pragma unroll
  for (int v = 0; v < 8; ++v) {
    int base = 2 * v;
    int k = base + 16 * lh;
    a0.h[2 * v]     = (_Float16)zrow[(size_t)k * S_DIM];
    a0.h[2 * v + 1] = (_Float16)zrow[(size_t)(k + 1) * S_DIM];
    a1.h[2 * v]     = (_Float16)zrow[(size_t)(k + 32) * S_DIM];
    a1.h[2 * v + 1] = (_Float16)zrow[(size_t)(k + 33) * S_DIM];
  }

  float best[8];
  int   bidx[8];
#pragma unroll
  for (int j = 0; j < 8; ++j) { best[j] = -3.4e38f; bidx[j] = 0; }

  // ---- score 512 codes: 8 iterations x 4 groups, 4 independent WMMA chains ----
  for (int g = 0; g < 32; g += 4) {
    V8F acc[4];
#pragma unroll
    for (int q = 0; q < 4; ++q) {
      const float bias = sBias[(g + q) * 16 + m];
#pragma unroll
      for (int j = 0; j < 8; ++j) acc[q].f[j] = bias;   // -0.5*s*|e|^2 via C operand
    }
#pragma unroll
    for (int q = 0; q < 4; ++q) {
      const v16h* pb = (const v16h*)&sB[(((g + q) * 2 + 0) * 32 + lane) * 16];
      acc[q].v = __builtin_amdgcn_wmma_f32_16x16x32_f16(false, a0.v, false, pb[0],
                                                        (short)0, acc[q].v, false, false);
    }
#pragma unroll
    for (int q = 0; q < 4; ++q) {
      const v16h* pb = (const v16h*)&sB[(((g + q) * 2 + 1) * 32 + lane) * 16];
      acc[q].v = __builtin_amdgcn_wmma_f32_16x16x32_f16(false, a1.v, false, pb[0],
                                                        (short)0, acc[q].v, false, false);
    }
#pragma unroll
    for (int q = 0; q < 4; ++q) {
      const int code = (g + q) * 16 + m;
#pragma unroll
      for (int j = 0; j < 8; ++j) {
        if (acc[q].f[j] > best[j]) { best[j] = acc[q].f[j]; bidx[j] = code; }  // strict > keeps first min
      }
    }
  }

  // ---- branchless argmax across the 16 columns of each half-wave ----
#define RED_STEP(XM)                                                         \
  {                                                                          \
    _Pragma("unroll") for (int j = 0; j < 8; ++j) {                          \
      float ov = swz_f<XM>(best[j]);                                         \
      int   oi = swz_i<XM>(bidx[j]);                                         \
      int take = (int)(ov > best[j]) | ((int)(ov == best[j]) & (int)(oi < bidx[j])); \
      best[j] = take ? ov : best[j];                                         \
      bidx[j] = take ? oi : bidx[j];                                         \
    }                                                                        \
  }
  RED_STEP(1) RED_STEP(2) RED_STEP(4) RED_STEP(8)
#undef RED_STEP

  if (m == 0) {
#pragma unroll
    for (int j = 0; j < 8; ++j) {
      int row = j + 8 * lh;                 // C-layout: lanes 0-15 -> M 0..7, lanes 16-31 -> M 8..15
      sIdx[wid * 16 + row] = bidx[j];
      out[IDX_OFF + n0 + row] = (float)bidx[j];
    }
  }
  __syncthreads();

  // ---- gather quantized rows, scatter to (B,C,T,H,W), accumulate loss ----
  const int code = sIdx[wid * 16 + m];
  const float* erow = emb + code * C_DIM;
  float* qb = out + (size_t)b * C_DIM * S_DIM;
  float lsum = 0.f;
  for (int cc = 0; cc < 32; ++cc) {
    int c = cc + 32 * lh;
    size_t off = (size_t)c * S_DIM + s0 + m;
    float q  = erow[c];
    float zv = zb[off];
    qb[off] = q;
    float d = q - zv;
    lsum += d * d;
  }
  lsum += swz_f<1>(lsum);
  lsum += swz_f<2>(lsum);
  lsum += swz_f<4>(lsum);
  lsum += swz_f<8>(lsum);
  lsum += swz_f<16>(lsum);
  if (lane == 0) atomicAdd(out + LOSS_OFF, lsum * (1.25f / (float)Q_ELEMS));
}

extern "C" void kernel_launch(void* const* d_in, const int* in_sizes, int n_in,
                              void* d_out, int out_size, void* d_ws, size_t ws_size,
                              hipStream_t stream) {
  (void)in_sizes; (void)n_in; (void)out_size; (void)d_ws; (void)ws_size;
  const float* z   = (const float*)d_in[0];
  const float* emb = (const float*)d_in[1];
  float* out = (float*)d_out;

  // zero the loss accumulator (graph-legal memset node)
  hipMemsetAsync((char*)d_out + (size_t)LOSS_OFF * sizeof(float), 0, sizeof(float), stream);

  // 16384 row-tiles, 4 waves (tiles) per 128-thread block
  vq_wmma_kernel<<<N_ROWS / 16 / 4, 128, 0, stream>>>(z, emb, out);
}
